// MRGSetAbstraction_46145128628931
// MI455X (gfx1250) — matrix operand order repaired and checked by reading, not compiled, not run on hardware
//
#include <hip/hip_runtime.h>
#include <hip/hip_bf16.h>

// MRG set-abstraction for MI455X (gfx1250, wave32, WMMA).
// Compute-bound MLP (~30 GFLOP) done with v_wmma_f32_16x16x32_f16, f32 accum.
// Gathered group tensors live only in LDS; HBM sees ~17MB total.

typedef __attribute__((ext_vector_type(16))) _Float16 v16h;
typedef __attribute__((ext_vector_type(8)))  _Float16 v8h;
typedef __attribute__((ext_vector_type(8)))  float    v8f;

#define BATCH 8
#define NPTS  4096
#define CIN   64
#define MCENT 1024      // N/4
#define KGRP  64
#define RAD2  0.04f     // 0.2^2
#define GPB   2         // centroid groups per block in mlp kernel (128 rows)

union AFrag { v16h v; v8h h[2]; };

// A-fragment (16x32 f16) from row-major LDS tile [rows][ld]:
// lane l, row = row0 + (l&15); halfs 0..15 = K[k0 + (l>>4)*16 ..]
// (contiguous K, matching load_b_glob)
__device__ __forceinline__ v16h load_a_lds(const _Float16* lds, int row0, int ld,
                                           int k0, int lane) {
  int r  = row0 + (lane & 15);
  int hh = lane >> 4;
  const _Float16* base = lds + r * ld + k0 + hh * 16;
  AFrag f;
  f.h[0] = *(const v8h*)(base);
  f.h[1] = *(const v8h*)(base + 8);
  return f.v;
}

// B-fragment (32x16 f16) from weights stored [Nout][Kdim] row-major (== torch (out,in)):
// lane l: column n = n0 + (l&15), 16 contiguous K halfs starting at k0 + (l>>4)*16.
__device__ __forceinline__ v16h load_b_glob(const _Float16* w, int k0, int n0,
                                            int ldk, int lane) {
  int n = n0 + (lane & 15);
  int k = k0 + (lane >> 4) * 16;
  return *(const v16h*)(w + (size_t)n * ldk + k);
}

// One MLP layer for this wave's 16-row M-tile. src is LDS act [.][ldsrc] f16,
// weights [NOUT][KDIM] f16 in global. A-fragments are loaded ONCE and held in
// VGPRs across all N-tiles (they are nt-invariant). Either writes relu(f16)
// act to dst LDS, or (TOMAX) reduces max over the tile's 16 rows into
// pm[mt*NOUT + col].
template<int KDIM, int NOUT, bool RELU, bool TOMAX>
__device__ __forceinline__ void layer(const _Float16* src, int ldsrc,
                                      const _Float16* wg, const float* bias,
                                      _Float16* dst, int lddst,
                                      float* pm, int mt, int lane)
{
  const int row0 = mt * 16;
  v16h afrag[KDIM / 32];
#pragma unroll
  for (int kk = 0; kk < KDIM / 32; ++kk)
    afrag[kk] = load_a_lds(src, row0, ldsrc, kk * 32, lane);

#pragma unroll
  for (int nt = 0; nt < NOUT / 16; ++nt) {
    v8f acc = {};
#pragma unroll
    for (int kk = 0; kk < KDIM / 32; ++kk) {
      v16h b = load_b_glob(wg, kk * 32, nt * 16, KDIM, lane);
      acc = __builtin_amdgcn_wmma_f32_16x16x32_f16(false, afrag[kk], false, b,
                                                   (short)0, acc, false, false);
    }
    const int   col = nt * 16 + (lane & 15);
    const float bv  = bias[col];
    if (TOMAX) {
      float mx = acc[0];
#pragma unroll
      for (int i = 1; i < 8; ++i) mx = fmaxf(mx, acc[i]);
      mx += bv;                                   // max then +b == max(v+b)
      mx = fmaxf(mx, __shfl_xor(mx, 16, 32));     // combine rows 0-7 with 8-15
      if (lane < 16) pm[mt * NOUT + col] = mx;
    } else {
      const int rbase = row0 + (lane >> 4) * 8;   // C layout: VGPR i -> row rbase+i
#pragma unroll
      for (int i = 0; i < 8; ++i) {
        float vv = acc[i] + bv;
        if (RELU) vv = vv > 0.f ? vv : 0.f;
        dst[(rbase + i) * lddst + col] = (_Float16)vv;
      }
    }
  }
}

// ---------------- prep kernels ----------------

__global__ void wprep_kernel(const float* __restrict__ fw1, const float* __restrict__ fw2,
                             const float* __restrict__ fw3, const float* __restrict__ pw1,
                             const float* __restrict__ pw2, const float* __restrict__ pw3,
                             _Float16* wF1, _Float16* wF2, _Float16* wF3,
                             _Float16* wP1, _Float16* wP2, _Float16* wP3)
{
  int t = blockIdx.x * blockDim.x + threadIdx.x;
  int stride = gridDim.x * blockDim.x;
  for (int e = t; e < 64 * 64;  e += stride) wF1[e] = (_Float16)fw1[e];
  for (int e = t; e < 64 * 64;  e += stride) wF2[e] = (_Float16)fw2[e];
  for (int e = t; e < 128 * 64; e += stride) wF3[e] = (_Float16)fw3[e];
  for (int e = t; e < 64 * 32;  e += stride) {           // pw1 [64][3] -> [64][32] K-padded
    int o = e >> 5, i = e & 31;
    wP1[e] = (_Float16)(i < 3 ? pw1[o * 3 + i] : 0.f);
  }
  for (int e = t; e < 64 * 64;  e += stride) wP2[e] = (_Float16)pw2[e];
  for (int e = t; e < 128 * 64; e += stride) wP3[e] = (_Float16)pw3[e];
}

__global__ void featprep_kernel(const float* __restrict__ pf, _Float16* featT)
{
  size_t total = (size_t)BATCH * NPTS * CIN;
  for (size_t e = (size_t)blockIdx.x * blockDim.x + threadIdx.x; e < total;
       e += (size_t)gridDim.x * blockDim.x) {
    size_t b = e / ((size_t)NPTS * CIN);
    size_t r = e % ((size_t)NPTS * CIN);
    size_t n = r / CIN, c = r % CIN;
    featT[e] = (_Float16)pf[(b * CIN + c) * NPTS + n];   // [B,C,N] -> [B,N,C] f16
  }
}

// ---------------- FPS (one block per batch) ----------------

__global__ __launch_bounds__(256) void fps_kernel(const float* __restrict__ points,
                                                  int* __restrict__ idxOut,
                                                  float* __restrict__ centOut)
{
  __shared__ float px[NPTS], py[NPTS], pz[NPTS];
  __shared__ float redV[256];
  __shared__ int   redI[256];
  const int b = blockIdx.x, tid = threadIdx.x;
  const float* P = points + (size_t)b * 3 * NPTS;
  for (int i = tid; i < NPTS; i += 256) { px[i] = P[i]; py[i] = P[NPTS + i]; pz[i] = P[2 * NPTS + i]; }
  __syncthreads();

  float mind[16];
#pragma unroll
  for (int j = 0; j < 16; ++j) mind[j] = 3.4e38f;
  int last = 0;
  for (int it = 0; it < MCENT; ++it) {
    if (tid == 0) {
      idxOut[b * MCENT + it] = last;
      float* c = centOut + ((size_t)b * MCENT + it) * 3;  // flat [B,M,3] == reference view
      c[0] = px[last]; c[1] = py[last]; c[2] = pz[last];
    }
    const float lx = px[last], ly = py[last], lz = pz[last];
    float best = -1.0f; int bi = 0x7fffffff;
#pragma unroll
    for (int j = 0; j < 16; ++j) {
      const int p = tid + j * 256;
      float dx = px[p] - lx, dy = py[p] - ly, dz = pz[p] - lz;
      float mv = fminf(mind[j], dx * dx + dy * dy + dz * dz);
      mind[j] = mv;
      if (mv > best) { best = mv; bi = p; }   // strict >: ascending p keeps first max
    }
    redV[tid] = best; redI[tid] = bi;
    __syncthreads();
    for (int s = 128; s > 0; s >>= 1) {
      if (tid < s) {
        float v2 = redV[tid + s]; int i2 = redI[tid + s];
        if (v2 > redV[tid] || (v2 == redV[tid] && i2 < redI[tid])) { redV[tid] = v2; redI[tid] = i2; }
      }
      __syncthreads();
    }
    last = redI[0];
    __syncthreads();
  }
}

// ---------------- ball query (one wave32 per centroid) ----------------

__global__ __launch_bounds__(256) void ball_kernel(const float* __restrict__ points,
                                                   const int* __restrict__ fpsidx,
                                                   int* __restrict__ sel)
{
  __shared__ int sbuf[8][KGRP];
  const int tid = threadIdx.x, lane = tid & 31, wave = tid >> 5;
  const int cm = blockIdx.x * 8 + wave;
  const int b  = cm / MCENT;
  const float* P = points + (size_t)b * 3 * NPTS;
  const int ci = fpsidx[cm];
  const float cx = P[ci], cy = P[NPTS + ci], cz = P[2 * NPTS + ci];

  int cnt = 0;
  for (int p0 = 0; p0 < NPTS && cnt < KGRP; p0 += 32) {
    const int p = p0 + lane;
    float dx = P[p] - cx, dy = P[NPTS + p] - cy, dz = P[2 * NPTS + p] - cz;
    bool in = (dx * dx + dy * dy + dz * dz) <= RAD2;
    unsigned m32 = (unsigned)__ballot(in);
    if (in) {
      int slot = cnt + __popc(m32 & ((1u << lane) - 1u));
      if (slot < KGRP) sbuf[wave][slot] = p;
    }
    cnt += __popc(m32);
  }
  if (cnt == 0) { if (lane == 0) sbuf[wave][0] = ci; cnt = 1; }   // centroid itself
  int first = sbuf[wave][0];
  for (int s = cnt + lane; s < KGRP; s += 32) sbuf[wave][s] = first;
  for (int s = lane; s < KGRP; s += 32) sel[(size_t)cm * KGRP + s] = sbuf[wave][s];
}

// ---------------- grouped MLPs + max-pool (WMMA) ----------------

__global__ __launch_bounds__(256) void mlp_kernel(
    const float* __restrict__ points, const int* __restrict__ fpsidx,
    const int* __restrict__ sel, const _Float16* __restrict__ featT,
    const _Float16* __restrict__ wF1, const _Float16* __restrict__ wF2,
    const _Float16* __restrict__ wF3, const _Float16* __restrict__ wP1,
    const _Float16* __restrict__ wP2, const _Float16* __restrict__ wP3,
    const float* __restrict__ fb1, const float* __restrict__ fb2,
    const float* __restrict__ fb3, const float* __restrict__ pb1,
    const float* __restrict__ pb2, const float* __restrict__ pb3,
    float* __restrict__ outFeat)
{
  __shared__ _Float16 actF[GPB * 64][64];   // gathered features (K-dim = 64)
  __shared__ _Float16 actP[GPB * 64][32];   // relative coords, K-padded 3 -> 32
  __shared__ _Float16 tmp [GPB * 64][64];   // layer ping-pong
  __shared__ float    pm  [GPB * 4][128];   // per-M-tile column maxes

  const int tid = threadIdx.x, lane = tid & 31, wave = tid >> 5;
  const int cm0 = blockIdx.x * GPB;

  __builtin_prefetch(wF1, 0, 0); __builtin_prefetch(wF2, 0, 0);
  __builtin_prefetch(wF3, 0, 0); __builtin_prefetch(wP1, 0, 0);
  __builtin_prefetch(wP2, 0, 0); __builtin_prefetch(wP3, 0, 0);

  // Gather features: 128 rows, 2 threads per row, 64B vector copies.
  {
    const int r = tid >> 1, hh = tid & 1;
    const int cm = cm0 + (r >> 6);
    const int b  = cm / MCENT;
    const int si = sel[(size_t)cm * KGRP + (r & 63)];
    const v8h* src = (const v8h*)(featT + ((size_t)b * NPTS + si) * CIN + hh * 32);
    v8h* dstp = (v8h*)(&actF[r][hh * 32]);
    dstp[0] = src[0]; dstp[1] = src[1]; dstp[2] = src[2]; dstp[3] = src[3];
  }
  // Relative coordinates (rows 0..127), zero-padded to K=32.
  if (tid < GPB * 64) {
    const int r  = tid;
    const int cm = cm0 + (r >> 6);
    const int b  = cm / MCENT;
    const int ci = fpsidx[cm];
    const float* P = points + (size_t)b * 3 * NPTS;
    const float cx = P[ci], cy = P[NPTS + ci], cz = P[2 * NPTS + ci];
    const int si = sel[(size_t)cm * KGRP + (r & 63)];
    v8h z = {};
    v8h* row = (v8h*)(&actP[r][0]);
    row[0] = z; row[1] = z; row[2] = z; row[3] = z;
    actP[r][0] = (_Float16)(P[si] - cx);
    actP[r][1] = (_Float16)(P[NPTS + si] - cy);
    actP[r][2] = (_Float16)(P[2 * NPTS + si] - cz);
  }
  __syncthreads();

  const int mt = wave;   // 8 waves, 8 M-tiles over 128 rows

  // feature net: 64 -> 64 -> 64 -> 128, relu, relu, (none) + max over K
  layer<64, 64, true,  false>(&actF[0][0], 64, wF1, fb1, &tmp[0][0],  64, nullptr,   mt, lane);
  __syncthreads();
  layer<64, 64, true,  false>(&tmp[0][0],  64, wF2, fb2, &actF[0][0], 64, nullptr,   mt, lane);
  __syncthreads();
  layer<64, 128, false, true>(&actF[0][0], 64, wF3, fb3, nullptr,      0, &pm[0][0], mt, lane);
  __syncthreads();
  {
    const int g = tid >> 7, c = tid & 127;
    float v = fmaxf(fmaxf(pm[g * 4 + 0][c], pm[g * 4 + 1][c]),
                    fmaxf(pm[g * 4 + 2][c], pm[g * 4 + 3][c]));
    const int cm = cm0 + g, b = cm / MCENT, m = cm % MCENT;
    outFeat[((size_t)b * 256 + c) * MCENT + m] = v;          // channels 0..127 = f_out
  }
  __syncthreads();

  // point net: 3(pad 32) -> 64 -> 64 -> 128, relu, relu, (none) + max over K
  layer<32, 64, true,  false>(&actP[0][0], 32, wP1, pb1, &tmp[0][0],  64, nullptr,   mt, lane);
  __syncthreads();
  layer<64, 64, true,  false>(&tmp[0][0],  64, wP2, pb2, &actF[0][0], 64, nullptr,   mt, lane);
  __syncthreads();
  layer<64, 128, false, true>(&actF[0][0], 64, wP3, pb3, nullptr,      0, &pm[0][0], mt, lane);
  __syncthreads();
  {
    const int g = tid >> 7, c = tid & 127;
    float v = fmaxf(fmaxf(pm[g * 4 + 0][c], pm[g * 4 + 1][c]),
                    fmaxf(pm[g * 4 + 2][c], pm[g * 4 + 3][c]));
    const int cm = cm0 + g, b = cm / MCENT, m = cm % MCENT;
    outFeat[((size_t)b * 256 + 128 + c) * MCENT + m] = v;    // channels 128..255 = p_out
  }
}

// ---------------- host ----------------

extern "C" void kernel_launch(void* const* d_in, const int* in_sizes, int n_in,
                              void* d_out, int out_size, void* d_ws, size_t ws_size,
                              hipStream_t stream) {
  const float* points = (const float*)d_in[0];
  const float* pfeat  = (const float*)d_in[1];
  const float* pw1 = (const float*)d_in[2];  const float* pb1 = (const float*)d_in[3];
  const float* pw2 = (const float*)d_in[4];  const float* pb2 = (const float*)d_in[5];
  const float* pw3 = (const float*)d_in[6];  const float* pb3 = (const float*)d_in[7];
  const float* fw1 = (const float*)d_in[8];  const float* fb1 = (const float*)d_in[9];
  const float* fw2 = (const float*)d_in[10]; const float* fb2 = (const float*)d_in[11];
  const float* fw3 = (const float*)d_in[12]; const float* fb3 = (const float*)d_in[13];

  float* out     = (float*)d_out;
  float* centOut = out;                                   // [B,M,3] flat == [B,3,M] view
  float* featOut = out + (size_t)BATCH * 3 * MCENT;       // [B,256,M]

  char* ws = (char*)d_ws;
  int*      idxw  = (int*)ws;                                          // 32 KB
  int*      selw  = (int*)(ws + (32 << 10));                           // 2 MB
  _Float16* featT = (_Float16*)(ws + (32 << 10) + (2 << 20));          // 4 MB
  _Float16* wF1   = (_Float16*)(ws + (32 << 10) + (2 << 20) + (4 << 20));
  _Float16* wF2 = wF1 + 64 * 64;
  _Float16* wF3 = wF2 + 64 * 64;
  _Float16* wP1 = wF3 + 128 * 64;
  _Float16* wP2 = wP1 + 64 * 32;
  _Float16* wP3 = wP2 + 64 * 64;

  wprep_kernel<<<16, 256, 0, stream>>>(fw1, fw2, fw3, pw1, pw2, pw3,
                                       wF1, wF2, wF3, wP1, wP2, wP3);
  featprep_kernel<<<512, 256, 0, stream>>>(pfeat, featT);
  fps_kernel<<<BATCH, 256, 0, stream>>>(points, idxw, centOut);
  ball_kernel<<<(BATCH * MCENT) / 8, 256, 0, stream>>>(points, idxw, selw);
  mlp_kernel<<<(BATCH * MCENT) / GPB, 256, 0, stream>>>(
      points, idxw, selw, featT, wF1, wF2, wF3, wP1, wP2, wP3,
      fb1, fb2, fb3, pb1, pb2, pb3, featOut);
}
